// MultiHeadQuasiLSTMCell_56461640073855
// MI455X (gfx1250) — matrix operand order repaired and checked by reading, not compiled, not run on hardware
//
#include <hip/hip_runtime.h>
#include <hip/hip_bf16.h>
#include <stdint.h>

// ---------------------------------------------------------------------------
// MultiHeadQuasiLSTM for MI455X (gfx1250, wave32, WMMA bf16 path)
//   B=4, S=4096, H=1024, NH=8, HD=128, L=64, d = NH*HD = 1024
// Round 3: native packed f32->bf16 conversion (__builtin_convertvector ->
// v_cvt_pk_bf16_f32) instead of integer RNE emulation; rest unchanged:
// double-buffered LDS, b128 fragment loads, rcp-based activations, NT streams.
// ---------------------------------------------------------------------------

#define B_    4
#define S_    4096
#define H_    1024
#define D_    1024          // NH*HD
#define ROWS  (B_ * S_)     // 16384
#define TK    32
#define LDSS  40            // LDS row stride (uint16): 80 B, 16B-aligned rows

typedef __attribute__((ext_vector_type(16))) __bf16 v16bf;
typedef __attribute__((ext_vector_type(8)))  float  v8f;
typedef __attribute__((ext_vector_type(2)))  float  v2f;
typedef __attribute__((ext_vector_type(2)))  __bf16 v2bf;

union FragBF { uint4 q[2]; uint32_t u[8]; v16bf v; };

__device__ __forceinline__ uint32_t pk2bf(float a, float b) {
  v2f vf = {a, b};
  union { v2bf b; uint32_t u; } cv;
  cv.b = __builtin_convertvector(vf, v2bf);    // native packed cvt on gfx1250
  return cv.u;
}

__device__ __forceinline__ uint16_t f2bf(float f) {
  union { __bf16 b; uint16_t u; } cv;
  cv.b = (__bf16)f;
  return cv.u;
}

__device__ __forceinline__ float sigm(float x) {
  return __builtin_amdgcn_rcpf(1.0f + __expf(-x));
}

__device__ __forceinline__ float tanh_fast(float x) {
  x = fminf(fmaxf(x, -10.0f), 10.0f);
  float e = __expf(2.0f * x);
  return (e - 1.0f) * __builtin_amdgcn_rcpf(e + 1.0f);
}

// ---------------------------------------------------------------------------
// Kernel 1: fused 4-gate GEMM  G[g] = act_g( X[g] (16384x1024) * W[g]^T + b[g] )
// blockIdx.z selects the gate. 128x128 C tile per block, 8 waves of 64x32.
// ---------------------------------------------------------------------------
__global__ void __launch_bounds__(256)
gates_gemm_kernel(const float* __restrict__ Xf, const float* __restrict__ Xi,
                  const float* __restrict__ Xz, const float* __restrict__ Xo,
                  const float* __restrict__ Wf, const float* __restrict__ Wi,
                  const float* __restrict__ Wz, const float* __restrict__ Wo,
                  const float* __restrict__ Bf, const float* __restrict__ Bi,
                  const float* __restrict__ Bz, const float* __restrict__ Bo,
                  float* __restrict__ Gf, float* __restrict__ Gi,
                  float* __restrict__ Gz, float* __restrict__ Go)
{
  const int g = blockIdx.z;
  const float* X  = (g == 0) ? Xf : (g == 1) ? Xi : (g == 2) ? Xz : Xo;
  const float* W  = (g == 0) ? Wf : (g == 1) ? Wi : (g == 2) ? Wz : Wo;
  const float* Bv = (g == 0) ? Bf : (g == 1) ? Bi : (g == 2) ? Bz : Bo;
  float*       G  = (g == 0) ? Gf : (g == 1) ? Gi : (g == 2) ? Gz : Go;

  __shared__ uint16_t sA[2][128][LDSS];
  __shared__ uint16_t sB[2][128][LDSS];

  const int tid  = threadIdx.x;
  const int lane = tid & 31;
  const int wid  = tid >> 5;
  const int wm   = wid >> 2;      // 0..1  -> 64-row slab
  const int wn   = wid & 3;       // 0..3  -> 32-col slab
  const int half = lane >> 4;
  const int l16  = lane & 15;
  const int M0   = blockIdx.x * 128;
  const int N0   = blockIdx.y * 128;

  // staging geometry: chunk i covers row i*32 + (tid>>3), cols (tid&7)*4
  const int ldr = tid >> 3;
  const int ldc = (tid & 7) << 2;
  const float* xbase = X + (size_t)(M0 + ldr) * H_ + ldc;
  const float* wbase = W + (size_t)(N0 + ldr) * H_ + ldc;

  v8f acc[4][2] = {};

  // ---- prologue: stage K-tile 0 into buffer 0 ----
#pragma unroll
  for (int i = 0; i < 4; ++i) {
    float4 va = *(const float4*)(xbase + (size_t)i * 32 * H_);
    float4 vb = *(const float4*)(wbase + (size_t)i * 32 * H_);
    *(uint2*)&sA[0][i * 32 + ldr][ldc] = make_uint2(pk2bf(va.x, va.y), pk2bf(va.z, va.w));
    *(uint2*)&sB[0][i * 32 + ldr][ldc] = make_uint2(pk2bf(vb.x, vb.y), pk2bf(vb.z, vb.w));
  }
  __syncthreads();

  const int NKT = H_ / TK;          // 32 K-steps
  for (int kt = 0; kt < NKT; ++kt) {
    const int cur = kt & 1;
    const int nxt = cur ^ 1;
    const bool have_next = (kt + 1) < NKT;

    // ---- issue global loads for K-tile kt+1 (latency hidden under WMMA) ----
    float4 ra[4], rb[4];
    if (have_next) {
      const int kn = (kt + 1) * TK;
#pragma unroll
      for (int i = 0; i < 4; ++i) {
        ra[i] = *(const float4*)(xbase + (size_t)i * 32 * H_ + kn);
        rb[i] = *(const float4*)(wbase + (size_t)i * 32 * H_ + kn);
      }
    }

    // ---- fragments from buffer `cur` (contiguous 16B chunks -> ds_load_b128)
    FragBF a[4], b[2];
#pragma unroll
    for (int t = 0; t < 4; ++t) {
      int row = wm * 64 + t * 16 + l16;
      a[t].q[0] = *(const uint4*)&sA[cur][row][half << 3];          // K 0-7 / 8-15
      a[t].q[1] = *(const uint4*)&sA[cur][row][(half << 3) + 16];   // K 16-23 / 24-31
    }
#pragma unroll
    for (int t = 0; t < 2; ++t) {
      int col = wn * 32 + t * 16 + l16;
      b[t].q[0] = *(const uint4*)&sB[cur][col][half << 3];
      b[t].q[1] = *(const uint4*)&sB[cur][col][(half << 3) + 16];
    }

#pragma unroll
    for (int mt = 0; mt < 4; ++mt)
#pragma unroll
      for (int nt = 0; nt < 2; ++nt)
        acc[mt][nt] = __builtin_amdgcn_wmma_f32_16x16x32_bf16(
            false, a[mt].v, false, b[nt].v, (short)0, acc[mt][nt], false, false);

    // ---- convert + store K-tile kt+1 into buffer `nxt` ----
    if (have_next) {
#pragma unroll
      for (int i = 0; i < 4; ++i) {
        *(uint2*)&sA[nxt][i * 32 + ldr][ldc] =
            make_uint2(pk2bf(ra[i].x, ra[i].y), pk2bf(ra[i].z, ra[i].w));
        *(uint2*)&sB[nxt][i * 32 + ldr][ldc] =
            make_uint2(pk2bf(rb[i].x, rb[i].y), pk2bf(rb[i].z, rb[i].w));
      }
    }
    __syncthreads();
  }

  // ---- epilogue: bias + gate activation, streaming fp32 out ----
#pragma unroll
  for (int nt = 0; nt < 2; ++nt) {
    int n = N0 + wn * 32 + nt * 16 + l16;
    float bias = Bv[n];
#pragma unroll
    for (int mt = 0; mt < 4; ++mt) {
      int mbase = M0 + wm * 64 + mt * 16 + half * 8;
      float* gp = G + (size_t)mbase * D_ + n;
#pragma unroll
      for (int r = 0; r < 8; ++r) {
        float v = acc[mt][nt][r] + bias;
        float out;
        if (g == 0)      out = sigm(v + 1.0f);   // forget gate
        else if (g == 2) out = tanh_fast(v);     // candidate
        else             out = sigm(v);          // input / output gates
        __builtin_nontemporal_store(out, gp);    // gates are stream-once (256MB > L2)
        gp += D_;
      }
    }
  }
}

// ---------------------------------------------------------------------------
// Kernel 2: first-order scan over time (4096 independent channels)
//   c = f*c + i*z ; h = o*tanh(c) ; H[s] = bf16(h)
// ---------------------------------------------------------------------------
__global__ void __launch_bounds__(256)
scan_kernel(const float* __restrict__ Gf, const float* __restrict__ Gi,
            const float* __restrict__ Gz, const float* __restrict__ Go,
            const float* __restrict__ c0, uint16_t* __restrict__ Hb,
            float* __restrict__ last_c, float* __restrict__ last_h)
{
  int idx = blockIdx.x * blockDim.x + threadIdx.x;   // 0..4095
  int b   = idx >> 10;
  int ch  = idx & 1023;
  size_t base = ((size_t)b * S_) * D_ + ch;

  float c = c0[idx];
  float h = 0.0f;
#pragma unroll 4
  for (int s = 0; s < S_; ++s) {
    size_t off = base + (size_t)s * D_;
    float f = __builtin_nontemporal_load(Gf + off);
    float i = __builtin_nontemporal_load(Gi + off);
    float z = __builtin_nontemporal_load(Gz + off);
    float o = __builtin_nontemporal_load(Go + off);
    c = __builtin_fmaf(f, c, i * z);
    h = o * tanh_fast(c);
    Hb[off] = f2bf(h);         // H is re-read by the projection: keep RT
  }
  last_c[idx] = c;
  last_h[idx] = h;
}

// ---------------------------------------------------------------------------
// Kernel 3: output projection  y = H(bf16, 16384x1024) * Wp^T + bp  (fp32 out)
// ---------------------------------------------------------------------------
__global__ void __launch_bounds__(256)
proj_gemm_kernel(const uint16_t* __restrict__ Hb, const float* __restrict__ Wp,
                 const float* __restrict__ bp, float* __restrict__ Y)
{
  __shared__ uint16_t sA[2][128][LDSS];
  __shared__ uint16_t sB[2][128][LDSS];

  const int tid  = threadIdx.x;
  const int lane = tid & 31;
  const int wid  = tid >> 5;
  const int wm   = wid >> 2;
  const int wn   = wid & 3;
  const int half = lane >> 4;
  const int l16  = lane & 15;
  const int M0   = blockIdx.x * 128;
  const int N0   = blockIdx.y * 128;

  // A staging (bf16 source): chunk i covers row i*64 + (tid>>2), 8-elem chunk tid&3
  const int ar = tid >> 2;
  const int ac = (tid & 3) << 3;
  const uint16_t* hbase = Hb + (size_t)(M0 + ar) * D_ + ac;
  // B staging (fp32 -> bf16): same geometry as gates kernel
  const int ldr = tid >> 3;
  const int ldc = (tid & 7) << 2;
  const float* wbase = Wp + (size_t)(N0 + ldr) * D_ + ldc;

  v8f acc[4][2] = {};

#pragma unroll
  for (int i = 0; i < 2; ++i)
    *(uint4*)&sA[0][i * 64 + ar][ac] = *(const uint4*)(hbase + (size_t)i * 64 * D_);
#pragma unroll
  for (int i = 0; i < 4; ++i) {
    float4 vb = *(const float4*)(wbase + (size_t)i * 32 * D_);
    *(uint2*)&sB[0][i * 32 + ldr][ldc] = make_uint2(pk2bf(vb.x, vb.y), pk2bf(vb.z, vb.w));
  }
  __syncthreads();

  const int NKT = D_ / TK;
  for (int kt = 0; kt < NKT; ++kt) {
    const int cur = kt & 1;
    const int nxt = cur ^ 1;
    const bool have_next = (kt + 1) < NKT;

    uint4  ha[2];
    float4 rb[4];
    if (have_next) {
      const int kn = (kt + 1) * TK;
#pragma unroll
      for (int i = 0; i < 2; ++i)
        ha[i] = *(const uint4*)(hbase + (size_t)i * 64 * D_ + kn);
#pragma unroll
      for (int i = 0; i < 4; ++i)
        rb[i] = *(const float4*)(wbase + (size_t)i * 32 * D_ + kn);
    }

    FragBF a[4], b[2];
#pragma unroll
    for (int t = 0; t < 4; ++t) {
      int row = wm * 64 + t * 16 + l16;
      a[t].q[0] = *(const uint4*)&sA[cur][row][half << 3];
      a[t].q[1] = *(const uint4*)&sA[cur][row][(half << 3) + 16];
    }
#pragma unroll
    for (int t = 0; t < 2; ++t) {
      int col = wn * 32 + t * 16 + l16;
      b[t].q[0] = *(const uint4*)&sB[cur][col][half << 3];
      b[t].q[1] = *(const uint4*)&sB[cur][col][(half << 3) + 16];
    }

#pragma unroll
    for (int mt = 0; mt < 4; ++mt)
#pragma unroll
      for (int nt = 0; nt < 2; ++nt)
        acc[mt][nt] = __builtin_amdgcn_wmma_f32_16x16x32_bf16(
            false, a[mt].v, false, b[nt].v, (short)0, acc[mt][nt], false, false);

    if (have_next) {
#pragma unroll
      for (int i = 0; i < 2; ++i)
        *(uint4*)&sA[nxt][i * 64 + ar][ac] = ha[i];
#pragma unroll
      for (int i = 0; i < 4; ++i)
        *(uint2*)&sB[nxt][i * 32 + ldr][ldc] = make_uint2(pk2bf(rb[i].x, rb[i].y),
                                                          pk2bf(rb[i].z, rb[i].w));
    }
    __syncthreads();
  }

#pragma unroll
  for (int nt = 0; nt < 2; ++nt) {
    int n = N0 + wn * 32 + nt * 16 + l16;
    float bias = bp[n];
#pragma unroll
    for (int mt = 0; mt < 4; ++mt) {
      int mbase = M0 + wm * 64 + mt * 16 + half * 8;
      float* yp = Y + (size_t)mbase * H_ + n;
#pragma unroll
      for (int r = 0; r < 8; ++r) {
        __builtin_nontemporal_store(acc[mt][nt][r] + bias, yp);  // y never re-read
        yp += H_;
      }
    }
  }
}

// ---------------------------------------------------------------------------
extern "C" void kernel_launch(void* const* d_in, const int* in_sizes, int n_in,
                              void* d_out, int out_size, void* d_ws, size_t ws_size,
                              hipStream_t stream) {
  (void)in_sizes; (void)n_in; (void)out_size; (void)ws_size;

  const float* f_in = (const float*)d_in[0];
  const float* i_in = (const float*)d_in[1];
  const float* z_in = (const float*)d_in[2];
  const float* o_in = (const float*)d_in[3];
  const float* c0   = (const float*)d_in[4];
  /* h0 = d_in[5] is unused by the math */
  const float* Wf = (const float*)d_in[6];
  const float* bf = (const float*)d_in[7];
  const float* Wi = (const float*)d_in[8];
  const float* bi = (const float*)d_in[9];
  const float* Wz = (const float*)d_in[10];
  const float* bz = (const float*)d_in[11];
  const float* Wo = (const float*)d_in[12];
  const float* bo = (const float*)d_in[13];
  const float* Wp = (const float*)d_in[14];
  const float* bp = (const float*)d_in[15];

  float* y      = (float*)d_out;                    // (B,S,H)
  float* last_c = y + (size_t)ROWS * H_;            // (B,NH,HD) = 4096
  float* last_h = last_c + (size_t)B_ * D_;         // 4096

  const size_t gsz = (size_t)ROWS * D_;
  float*    gF = (float*)d_ws;
  float*    gI = gF + gsz;
  float*    gZ = gI + gsz;
  float*    gO = gZ + gsz;
  uint16_t* Hb = (uint16_t*)(gO + gsz);             // bf16 H, 32 MB

  dim3 grid1(ROWS / 128, D_ / 128, 4);
  gates_gemm_kernel<<<grid1, 256, 0, stream>>>(
      f_in, i_in, z_in, o_in, Wf, Wi, Wz, Wo, bf, bi, bz, bo, gF, gI, gZ, gO);

  scan_kernel<<<(B_ * D_) / 256, 256, 0, stream>>>(
      gF, gI, gZ, gO, c0, Hb, last_c, last_h);

  dim3 grid2(ROWS / 128, H_ / 128);
  proj_gemm_kernel<<<grid2, 256, 0, stream>>>(Hb, Wp, bp, y);
}
